// Attention_49177375539248
// MI455X (gfx1250) — matrix operand order repaired and checked by reading, not compiled, not run on hardware
//
#include <hip/hip_runtime.h>

// ---- CDNA5 WMMA / TDM types ------------------------------------------------
typedef __attribute__((ext_vector_type(16))) __bf16 v16bf;
typedef __attribute__((ext_vector_type(8)))  __bf16 v8bf;
typedef __attribute__((ext_vector_type(8)))  float  v8f;
typedef __attribute__((ext_vector_type(4)))  unsigned int u32x4;
typedef __attribute__((ext_vector_type(8)))  int i32x8;
typedef __attribute__((ext_vector_type(4)))  int i32x4;

#define DIM     768
#define NHEAD   12
#define HDIM    64
#define SEQ     1024
#define BATCH   8
#define TOKENS  (BATCH * SEQ)
#define QKV_OUT (3 * DIM)
#define SCALE   0.125f   // HEAD_DIM^-0.5 = 1/8

// low 32 bits of a generic pointer to __shared__ = LDS byte offset
#define LDSOFF(p) ((unsigned)(unsigned long long)(p))

// fp32 -> bf16, round-to-nearest-even
static __device__ __forceinline__ __bf16 f2bf(float f) {
  unsigned u = __builtin_bit_cast(unsigned, f);
  u += 0x7FFFu + ((u >> 16) & 1u);
  unsigned short h = (unsigned short)(u >> 16);
  return __builtin_bit_cast(__bf16, h);
}

static __device__ __forceinline__ v8f wmma_bf16(v16bf a, v16bf b, v8f c) {
  return __builtin_amdgcn_wmma_f32_16x16x32_bf16(false, a, false, b, (short)0, c,
                                                 false, false);
}

// 16x32 bf16 fragment assembled from two 16B runs
union Frag { v16bf v; v8bf h[2]; };

// ---- Tensor Data Mover: 2-D bf16 tile -> LDS (ISA ch.8 D# layout) ----------
static __device__ __forceinline__ void
tdm_load_2d(unsigned lds_off, const void* gptr,
            unsigned tensor_d0, unsigned tensor_d1,
            unsigned tile_d0, unsigned tile_d1, unsigned long stride0) {
  unsigned long ga = (unsigned long)gptr;
  u32x4 g0;
  g0[0] = 1u;                                   // count=1, user mode, no gather
  g0[1] = lds_off;                              // lds_addr
  g0[2] = (unsigned)(ga & 0xffffffffu);         // global_addr[31:0]
  g0[3] = (unsigned)((ga >> 32) & 0x1ffffffu)   // global_addr[56:32]
        | (2u << 30);                           // type = 2 ("image")
  i32x8 g1;
  g1[0] = (int)(1u << 16);                      // workgroup_mask=0, data_size=1 (2B)
  g1[1] = (int)((tensor_d0 & 0xffffu) << 16);   // tensor_dim0[15:0] @bits63:48
  g1[2] = (int)((tensor_d0 >> 16) | ((tensor_d1 & 0xffffu) << 16));
  g1[3] = (int)((tensor_d1 >> 16) | (tile_d0 << 16));   // tile_dim0 @127:112
  g1[4] = (int)(tile_d1 & 0xffffu);             // tile_dim1; tile_dim2=0
  g1[5] = (int)(stride0 & 0xffffffffu);         // tensor_dim0_stride[31:0]
  g1[6] = (int)((stride0 >> 32) & 0xffffu);     // stride0[47:32]; dim1_stride=0
  g1[7] = 0;
  i32x4 z4 = {0, 0, 0, 0};                      // groups 2/3 unused (2-D tile)
  i32x8 z8 = {0, 0, 0, 0, 0, 0, 0, 0};
  __builtin_amdgcn_tensor_load_to_lds(g0, g1, z4, z4, z8, 0);
}

// ---- stage 0: fp32 -> bf16 conversion --------------------------------------
__global__ void cvt_f32_bf16(const float* __restrict__ src,
                             __bf16* __restrict__ dst, int n) {
  for (int i = blockIdx.x * blockDim.x + threadIdx.x; i < n;
       i += gridDim.x * blockDim.x)
    dst[i] = f2bf(src[i]);
}

// ---- stage 1: QKV GEMM -----------------------------------------------------
// Wave tile 32x64: 2 A-frags x 4 B-frags -> 8 WMMAs per 32-deep k-step.
__global__ void __launch_bounds__(256)
qkv_gemm(const __bf16* __restrict__ X, const __bf16* __restrict__ W,
         __bf16* __restrict__ Qb, __bf16* __restrict__ Kb,
         __bf16* __restrict__ Vt) {
  const int lane = threadIdx.x & 31, wave = threadIdx.x >> 5;
  const int l16 = lane & 15, g = lane >> 4;
  const int m0 = blockIdx.y * 256 + wave * 32;
  const int n0 = blockIdx.x * 64;
  const __bf16* xrow0 = X + (size_t)(m0 + l16) * DIM;
  const __bf16* xrow1 = X + (size_t)(m0 + 16 + l16) * DIM;

  v8f acc[2][4] = {};
  for (int kk = 0; kk < DIM; kk += 32) {
    Frag a0, a1;
    a0.h[0] = *(const v8bf*)(xrow0 + kk + 16 * g);
    a0.h[1] = *(const v8bf*)(xrow0 + kk + 16 * g + 8);
    a1.h[0] = *(const v8bf*)(xrow1 + kk + 16 * g);
    a1.h[1] = *(const v8bf*)(xrow1 + kk + 16 * g + 8);
#pragma unroll
    for (int nc = 0; nc < 4; ++nc) {
      v16bf bf = *(const v16bf*)(W + (size_t)(n0 + nc * 16 + l16) * DIM + kk + 16 * g);
      acc[0][nc] = wmma_bf16(a0.v, bf, acc[0][nc]);
      acc[1][nc] = wmma_bf16(a1.v, bf, acc[1][nc]);
    }
  }
#pragma unroll
  for (int mi = 0; mi < 2; ++mi) {
#pragma unroll
    for (int nc = 0; nc < 4; ++nc) {
      const int ocol = n0 + nc * 16 + l16;
      const int three = ocol / DIM, rem = ocol % DIM;
      const int h = rem / HDIM, d = rem % HDIM;
#pragma unroll
      for (int r = 0; r < 8; ++r) {
        const int tok = m0 + mi * 16 + r + 8 * g;
        const int b = tok >> 10, nt = tok & (SEQ - 1);
        const int bh = b * NHEAD + h;
        const float v = acc[mi][nc][r];
        if (three == 0)
          Qb[((size_t)bh * SEQ + nt) * HDIM + d] = f2bf(v * SCALE);
        else if (three == 1)
          Kb[((size_t)bh * SEQ + nt) * HDIM + d] = f2bf(v);
        else
          Vt[((size_t)bh * HDIM + d) * SEQ + nt] = f2bf(v);
      }
    }
  }
}

// ---- stage 2: flash attention with TDM-staged K/V --------------------------
// Block = 8 waves x 16 query rows. TDM double-buffers 64-key chunks of
// K [64x64] and V^T [64x64] into LDS; waves consume them as WMMA B-frags.
__global__ void __launch_bounds__(256)
flash_attn(const __bf16* __restrict__ Qb, const __bf16* __restrict__ Kb,
           const __bf16* __restrict__ Vt, __bf16* __restrict__ O) {
  __shared__ __align__(32) __bf16 Ksh[2][64 * 64];   // [key][d]
  __shared__ __align__(32) __bf16 Vsh[2][64 * 64];   // [d][key]
  __shared__ __align__(32) __bf16 plds[8][16 * 40];  // P staging, padded rows
  const int lane = threadIdx.x & 31, wave = threadIdx.x >> 5;
  const int l16 = lane & 15, g = lane >> 4;
  const int b = blockIdx.z, h = blockIdx.y, bh = b * NHEAD + h;
  const int q0 = blockIdx.x * 128 + wave * 16;

  const __bf16* qrow = Qb + ((size_t)bh * SEQ + q0 + l16) * HDIM;
  Frag qa[2];
#pragma unroll
  for (int dc = 0; dc < 2; ++dc) {
    qa[dc].h[0] = *(const v8bf*)(qrow + 32 * dc + 16 * g);
    qa[dc].h[1] = *(const v8bf*)(qrow + 32 * dc + 16 * g + 8);
  }
  const __bf16* kglob = Kb + (size_t)bh * SEQ * HDIM;   // + kt*HDIM
  const __bf16* vglob = Vt + (size_t)bh * HDIM * SEQ;   // + d*SEQ + kt
  __bf16* pl = &plds[wave][0];

  v8f o0 = {}, o1 = {}, o2 = {}, o3 = {};
  float mrow[8], lrow[8];
#pragma unroll
  for (int r = 0; r < 8; ++r) { mrow[r] = -1e30f; lrow[r] = 0.f; }

  if (wave == 0) {  // prologue DMA: chunk 0
    tdm_load_2d(LDSOFF(&Ksh[0][0]), kglob, 64, 65536u, 64, 64, 64);
    tdm_load_2d(LDSOFF(&Vsh[0][0]), vglob, SEQ, 65536u, 64, 64, SEQ);
  }

  for (int c = 0; c < SEQ / 64; ++c) {
    const int kt = c * 64, buf = c & 1;
    if (wave == 0) {
      if (c + 1 < SEQ / 64) {   // issue next chunk into the other buffer
        const int nb = buf ^ 1, nkt = kt + 64;
        tdm_load_2d(LDSOFF(&Ksh[nb][0]), kglob + (size_t)nkt * HDIM, 64, 65536u, 64, 64, 64);
        tdm_load_2d(LDSOFF(&Vsh[nb][0]), vglob + nkt, SEQ, 65536u, 64, 64, SEQ);
        __builtin_amdgcn_s_wait_tensorcnt((short)2);  // current chunk complete
      } else {
        __builtin_amdgcn_s_wait_tensorcnt((short)0);
      }
    }
    __syncthreads();                       // publish Ksh/Vsh[buf]
    const __bf16* kb = &Ksh[buf][0];
    const __bf16* vb = &Vsh[buf][0];

#pragma unroll
    for (int sub = 0; sub < 2; ++sub) {    // two 32-key subtiles per chunk
      const int ko = sub * 32;
      v8f s0 = {}, s1 = {};
      {
        Frag b0, b1;                       // keys ko+l16, d-chunks 0 / 1
        const __bf16* p = kb + (ko + l16) * 64 + 16 * g;
        b0.h[0] = *(const v8bf*)(p);       b0.h[1] = *(const v8bf*)(p + 8);
        b1.h[0] = *(const v8bf*)(p + 32);  b1.h[1] = *(const v8bf*)(p + 40);
        s0 = wmma_bf16(qa[0].v, b0.v, s0);
        s0 = wmma_bf16(qa[1].v, b1.v, s0);
        const __bf16* q = kb + (ko + 16 + l16) * 64 + 16 * g;
        b0.h[0] = *(const v8bf*)(q);       b0.h[1] = *(const v8bf*)(q + 8);
        b1.h[0] = *(const v8bf*)(q + 32);  b1.h[1] = *(const v8bf*)(q + 40);
        s1 = wmma_bf16(qa[0].v, b0.v, s1);
        s1 = wmma_bf16(qa[1].v, b1.v, s1);
      }
      // online softmax (rows live in 16-lane halves; xor<16 stays inside)
#pragma unroll
      for (int r = 0; r < 8; ++r) {
        float t = fmaxf(s0[r], s1[r]);
        t = fmaxf(t, __shfl_xor(t, 1));
        t = fmaxf(t, __shfl_xor(t, 2));
        t = fmaxf(t, __shfl_xor(t, 4));
        t = fmaxf(t, __shfl_xor(t, 8));
        const float mold = mrow[r];
        const float mnew = fmaxf(mold, t);
        const float alpha = __expf(mold - mnew);
        const float p0 = __expf(s0[r] - mnew);
        const float p1 = __expf(s1[r] - mnew);
        float rs = p0 + p1;
        rs += __shfl_xor(rs, 1);
        rs += __shfl_xor(rs, 2);
        rs += __shfl_xor(rs, 4);
        rs += __shfl_xor(rs, 8);
        lrow[r] = lrow[r] * alpha + rs;
        mrow[r] = mnew;
        o0[r] *= alpha; o1[r] *= alpha; o2[r] *= alpha; o3[r] *= alpha;
        const int m = r + 8 * g;
        pl[m * 40 + l16]      = f2bf(p0);
        pl[m * 40 + 16 + l16] = f2bf(p1);
      }
      asm volatile("" ::: "memory");       // LDS is in-order per wave
      Frag pa;
      pa.h[0] = *(const v8bf*)(pl + l16 * 40 + 16 * g);
      pa.h[1] = *(const v8bf*)(pl + l16 * 40 + 16 * g + 8);
      // O += P * V from LDS V^T [d][key]
#pragma unroll
      for (int cn = 0; cn < 4; ++cn) {
        Frag vbf;
        const __bf16* vp = vb + (cn * 16 + l16) * 64 + ko + 16 * g;
        vbf.h[0] = *(const v8bf*)(vp);
        vbf.h[1] = *(const v8bf*)(vp + 8);
        if (cn == 0)      o0 = wmma_bf16(pa.v, vbf.v, o0);
        else if (cn == 1) o1 = wmma_bf16(pa.v, vbf.v, o1);
        else if (cn == 2) o2 = wmma_bf16(pa.v, vbf.v, o2);
        else              o3 = wmma_bf16(pa.v, vbf.v, o3);
      }
    }
    __syncthreads();   // everyone done with buf before it is overwritten
  }
  // normalize, write [B,N,H*D] bf16
#pragma unroll
  for (int r = 0; r < 8; ++r) {
    const float inv = 1.f / lrow[r];
    const int tok = q0 + r + 8 * g;
    __bf16* orow = O + ((size_t)b * SEQ + tok) * DIM + h * HDIM;
    orow[ 0 + l16] = f2bf(o0[r] * inv);
    orow[16 + l16] = f2bf(o1[r] * inv);
    orow[32 + l16] = f2bf(o2[r] * inv);
    orow[48 + l16] = f2bf(o3[r] * inv);
  }
}

// ---- stage 3: output projection + bias (f32 out) ---------------------------
__global__ void __launch_bounds__(256)
proj_gemm(const __bf16* __restrict__ A, const __bf16* __restrict__ W,
          const float* __restrict__ bias, float* __restrict__ out) {
  const int lane = threadIdx.x & 31, wave = threadIdx.x >> 5;
  const int l16 = lane & 15, g = lane >> 4;
  const int m0 = blockIdx.y * 256 + wave * 32;
  const int n0 = blockIdx.x * 64;
  const __bf16* arow0 = A + (size_t)(m0 + l16) * DIM;
  const __bf16* arow1 = A + (size_t)(m0 + 16 + l16) * DIM;

  v8f acc[2][4] = {};
  for (int kk = 0; kk < DIM; kk += 32) {
    Frag a0, a1;
    a0.h[0] = *(const v8bf*)(arow0 + kk + 16 * g);
    a0.h[1] = *(const v8bf*)(arow0 + kk + 16 * g + 8);
    a1.h[0] = *(const v8bf*)(arow1 + kk + 16 * g);
    a1.h[1] = *(const v8bf*)(arow1 + kk + 16 * g + 8);
#pragma unroll
    for (int nc = 0; nc < 4; ++nc) {
      v16bf bf = *(const v16bf*)(W + (size_t)(n0 + nc * 16 + l16) * DIM + kk + 16 * g);
      acc[0][nc] = wmma_bf16(a0.v, bf, acc[0][nc]);
      acc[1][nc] = wmma_bf16(a1.v, bf, acc[1][nc]);
    }
  }
#pragma unroll
  for (int mi = 0; mi < 2; ++mi) {
#pragma unroll
    for (int nc = 0; nc < 4; ++nc) {
      const int ocol = n0 + nc * 16 + l16;
      const float bv = bias[ocol];
#pragma unroll
      for (int r = 0; r < 8; ++r) {
        const int tok = m0 + mi * 16 + r + 8 * g;
        out[(size_t)tok * DIM + ocol] = acc[mi][nc][r] + bv;
      }
    }
  }
}

// ---- launch ----------------------------------------------------------------
extern "C" void kernel_launch(void* const* d_in, const int* in_sizes, int n_in,
                              void* d_out, int out_size, void* d_ws, size_t ws_size,
                              hipStream_t stream) {
  const float* x      = (const float*)d_in[0];
  const float* qkv_w  = (const float*)d_in[1];
  const float* proj_w = (const float*)d_in[2];
  const float* proj_b = (const float*)d_in[3];
  float* out = (float*)d_out;

  char* p = (char*)d_ws;
  __bf16* xb   = (__bf16*)p; p += (size_t)TOKENS  * DIM * 2;
  __bf16* wqkv = (__bf16*)p; p += (size_t)QKV_OUT * DIM * 2;
  __bf16* wp   = (__bf16*)p; p += (size_t)DIM     * DIM * 2;
  const size_t bhnd = (size_t)BATCH * NHEAD * SEQ * HDIM * 2;
  __bf16* Qb = (__bf16*)p; p += bhnd;
  __bf16* Kb = (__bf16*)p; p += bhnd;
  __bf16* Vt = (__bf16*)p; p += bhnd;   // V stored transposed [B,H,D,N]
  __bf16* Ob = (__bf16*)p; p += (size_t)TOKENS * DIM * 2;

  cvt_f32_bf16<<<1024, 256, 0, stream>>>(x,      xb,   TOKENS * DIM);
  cvt_f32_bf16<<<512,  256, 0, stream>>>(qkv_w,  wqkv, QKV_OUT * DIM);
  cvt_f32_bf16<<<256,  256, 0, stream>>>(proj_w, wp,   DIM * DIM);

  qkv_gemm<<<dim3(QKV_OUT / 64, TOKENS / 256), 256, 0, stream>>>(xb, wqkv, Qb, Kb, Vt);
  flash_attn<<<dim3(SEQ / 128, NHEAD, BATCH), 256, 0, stream>>>(Qb, Kb, Vt, Ob);
  proj_gemm<<<dim3(DIM / 64, TOKENS / 256), 256, 0, stream>>>(Ob, wp, proj_b, out);
}
